// SRModel_54958401519987
// MI455X (gfx1250) — matrix operand order repaired and checked by reading, not compiled, not run on hardware
//
#include <hip/hip_runtime.h>

// ---------------------------------------------------------------------------
// Types for CDNA5 WMMA (wave32)
// ---------------------------------------------------------------------------
typedef __attribute__((ext_vector_type(8)))  unsigned short v8us;
typedef __attribute__((ext_vector_type(16))) unsigned short v16us;
typedef __attribute__((ext_vector_type(16))) __bf16         v16bf;
typedef __attribute__((ext_vector_type(8)))  float          v8f;

#define NB        2
#define LRH       64
#define LRW       64
#define HRH       256
#define HRW       256
#define FEATC     64
#define HID       256
#define KIN_PAD   128      // 106 -> 128
#define NHEAD_PAD 224      // 196+4 -> 224
#define MROWS     64       // HR pixels per block
// LDS: actA (64*256 bf16) | region2 = actB (bf16) U headbuf (64*224 f32) | weight stage
#define SMEM_ACTA   (MROWS*HID*2)          // 32768
#define SMEM_REG2   (MROWS*NHEAD_PAD*4)    // 57344
#define SMEM_WBUF   (HID*HID*2)            // 131072 (largest layer weight)
#define SMEM_BYTES  (SMEM_ACTA + SMEM_REG2 + SMEM_WBUF)   // 221184

__device__ __forceinline__ int   iclamp(int v, int lo, int hi){ return v<lo?lo:(v>hi?hi:v); }
__device__ __forceinline__ float fclampf(float v,float lo,float hi){ return fminf(fmaxf(v,lo),hi); }

__device__ __forceinline__ unsigned short f2bf(float f){
  unsigned int u = __float_as_uint(f);
  u += 0x7FFFu + ((u>>16)&1u);            // round-to-nearest-even
  return (unsigned short)(u>>16);
}

// Branchless GELU (tanh form): v_exp_f32 + v_rcp_f32, no EXEC-mask branches,
// so it co-executes cleanly with XDL WMMA.
__device__ __forceinline__ float gelu_f(float x){
  float z  = 0.7978845608028654f * (x + 0.044715f * x * x * x);
  float e  = __builtin_amdgcn_exp2f(z * 2.8853900817779268f);   // e^{2z}
  float th = 1.0f - 2.0f * __builtin_amdgcn_rcpf(e + 1.0f);     // tanh(z)
  return 0.5f * x * (1.0f + th);
}

// A-fragment (16-bit, 16x32): lane holds row M=lane%16; K = half*16 + [0..15] contiguous
__device__ __forceinline__ v16bf load_afrag(const unsigned short* p){
  v8us lo = *(const v8us*)(p);
  v8us hi = *(const v8us*)(p + 8);
  v16us c = __builtin_shufflevector(lo, hi, 0,1,2,3,4,5,6,7,8,9,10,11,12,13,14,15);
  return __builtin_bit_cast(v16bf, c);
}
// B-fragment (16-bit, 32x16): lane holds col N=lane%16; K = (lane/16)*16 + [0..15] contiguous
__device__ __forceinline__ v16bf load_bfrag(const unsigned short* p){
  v16us c = *(const v16us*)(p);
  return __builtin_bit_cast(v16bf, c);
}

// ---------------------------------------------------------------------------
// Async weight staging: global bf16 [N][K] -> LDS, via gfx1250 async-to-LDS
// path (ASYNCcnt). 256 threads x 16B per issue = 4KB/issue.
// ---------------------------------------------------------------------------
__device__ __forceinline__ void async_issue_weights(const unsigned short* __restrict__ Wt,
                                                    unsigned short* wlds, int bytes, int tid)
{
  unsigned int      l = (unsigned int)(uintptr_t)wlds + (unsigned int)tid*16u;
  unsigned long long g = (unsigned long long)(uintptr_t)Wt + (unsigned long long)tid*16u;
  for (int off = tid*16; off < bytes; off += 256*16) {
    asm volatile("global_load_async_to_lds_b128 %0, %1, off"
                 :: "v"(l), "v"(g) : "memory");
    l += 4096u;
    g += 4096ull;
  }
}
__device__ __forceinline__ void async_wait_all(){
  asm volatile("s_wait_asynccnt 0x0" ::: "memory");
}

// ---------------------------------------------------------------------------
// WMMA GEMM tiles: X[64][HID] (LDS bf16, ld=HID) x W (LDS bf16 [N][K])
// waves: mt = wave&3 selects 16-row M tile, nh = wave>>2 selects N half.
// A-fragments are hoisted into registers and reused across all N tiles.
// ---------------------------------------------------------------------------
template<int K>
__device__ __forceinline__ void gemm_layer(const unsigned short* Xs,
                                           const unsigned short* Wl,     // LDS staged weights
                                           const float* __restrict__ bias,
                                           unsigned short* Ys,
                                           int lane, int mt, int nh)
{
  const int l15  = lane & 15;
  const int half = lane >> 4;
  const unsigned short* xrow = Xs + (mt*16 + l15)*HID + half*16;
  v16bf afrag[K/32];
#pragma unroll
  for (int kk = 0; kk < K/32; ++kk) afrag[kk] = load_afrag(xrow + kk*32);

  for (int ntl = 0; ntl < 8; ++ntl) {
    const int nbase = (nh*8 + ntl)*16;
    const unsigned short* wrow = Wl + (nbase + l15)*K + half*16;
    v8f acc = {0.f,0.f,0.f,0.f,0.f,0.f,0.f,0.f};
#pragma unroll
    for (int kk = 0; kk < K/32; ++kk) {
      v16bf b = load_bfrag(wrow + kk*32);
      acc = __builtin_amdgcn_wmma_f32_16x16x32_bf16(false, afrag[kk], false, b,
                                                    (short)0, acc, false, false);
    }
    const float bn = bias[nbase + l15];
#pragma unroll
    for (int r = 0; r < 8; ++r) {
      float v = gelu_f(acc[r] + bn);
      Ys[(mt*16 + half*8 + r)*HID + nbase + l15] = f2bf(v);
    }
  }
}

__device__ __forceinline__ void gemm_head(const unsigned short* Xs,
                                          const unsigned short* Wl,
                                          const float* __restrict__ bias,
                                          float* Ys,
                                          int lane, int mt, int nh)
{
  const int l15  = lane & 15;
  const int half = lane >> 4;
  const unsigned short* xrow = Xs + (mt*16 + l15)*HID + half*16;
  v16bf afrag[HID/32];
#pragma unroll
  for (int kk = 0; kk < HID/32; ++kk) afrag[kk] = load_afrag(xrow + kk*32);

  for (int ntl = 0; ntl < 7; ++ntl) {                  // 14 N-tiles total = 224
    const int nbase = (nh*7 + ntl)*16;
    const unsigned short* wrow = Wl + (nbase + l15)*HID + half*16;
    v8f acc = {0.f,0.f,0.f,0.f,0.f,0.f,0.f,0.f};
#pragma unroll
    for (int kk = 0; kk < HID/32; ++kk) {
      v16bf b = load_bfrag(wrow + kk*32);
      acc = __builtin_amdgcn_wmma_f32_16x16x32_bf16(false, afrag[kk], false, b,
                                                    (short)0, acc, false, false);
    }
    const float bn = bias[nbase + l15];
#pragma unroll
    for (int r = 0; r < 8; ++r)
      Ys[(mt*16 + half*8 + r)*NHEAD_PAD + nbase + l15] = acc[r] + bn;
  }
}

// ---------------------------------------------------------------------------
// Weight prep: f32 [K][N] -> bf16 [N][Kpad]
// ---------------------------------------------------------------------------
__global__ void k_transpose_bf16(const float* __restrict__ w, unsigned short* __restrict__ wt,
                                 int K, int N, int Kpad)
{
  int idx = blockIdx.x*blockDim.x + threadIdx.x;
  if (idx >= N*Kpad) return;
  int n = idx / Kpad, k = idx % Kpad;
  float v = (k < K) ? w[(size_t)k*N + n] : 0.f;
  wt[idx] = f2bf(v);
}

__global__ void k_prep_head(const float* __restrict__ hkw, const float* __restrict__ hkb,
                            const float* __restrict__ hgw, const float* __restrict__ hgb,
                            unsigned short* __restrict__ wt, float* __restrict__ hbias)
{
  int idx = blockIdx.x*blockDim.x + threadIdx.x;
  if (idx >= NHEAD_PAD*HID) return;
  int n = idx / HID, k = idx % HID;
  float v = 0.f;
  if (n < 196)      v = hkw[(size_t)k*196 + n];
  else if (n < 200) v = hgw[(size_t)k*4 + (n-196)];
  wt[(size_t)n*HID + k] = f2bf(v);
  if (k == 0)
    hbias[n] = (n < 196) ? hkb[n] : ((n < 200) ? hgb[n-196] : 0.f);
}

// ---------------------------------------------------------------------------
// Encoder conv 3x3 SAME + GELU (fp32, output channels-last [B][H][W][64])
// ---------------------------------------------------------------------------
__global__ void k_conv_enc(const float* __restrict__ in, int cin, int in_nchw,
                           const float* __restrict__ w, const float* __restrict__ bias,
                           float* __restrict__ out)
{
  int idx = blockIdx.x*blockDim.x + threadIdx.x;
  if (idx >= NB*LRH*LRW*FEATC) return;
  int oc = idx & 63;
  int t  = idx >> 6;
  int x = t & 63; t >>= 6;
  int y = t & 63;
  int b = t >> 6;
  float s = bias[oc];
  for (int ky = 0; ky < 3; ++ky) {
    int iy = y + ky - 1;
    if (iy < 0 || iy >= LRH) continue;
    for (int kx = 0; kx < 3; ++kx) {
      int ix = x + kx - 1;
      if (ix < 0 || ix >= LRW) continue;
      const float* wp = w + (size_t)oc*cin*9 + ky*3 + kx;
      if (in_nchw) {
        const float* ip = in + ((size_t)(b*cin)*LRH + iy)*LRW + ix;
        for (int ic = 0; ic < cin; ++ic) s += ip[(size_t)ic*LRH*LRW] * wp[ic*9];
      } else {
        const float* ip = in + (((size_t)(b*LRH) + iy)*LRW + ix)*cin;
        for (int ic = 0; ic < cin; ++ic) s += ip[ic] * wp[ic*9];
      }
    }
  }
  out[(((size_t)(b*LRH)+y)*LRW + x)*FEATC + oc] = gelu_f(s);
}

// ---------------------------------------------------------------------------
// Fused per-pixel MLP + heads + patch mixing  ->  res0 (NCHW)
// ---------------------------------------------------------------------------
__global__ void __launch_bounds__(256)
k_mlp_fused(const float* __restrict__ feat,   // [B][64][64][64]
            const float* __restrict__ lr,     // [B][3][64][64]
            const unsigned short* W0, const float* b0,
            const unsigned short* W1, const float* b1,
            const unsigned short* W2, const float* b2,
            const unsigned short* W3, const float* b3,
            const unsigned short* W4, const float* b4,
            const unsigned short* WH, const float* bH,
            float* __restrict__ res0)
{
  extern __shared__ char smem[];
  unsigned short* bufA    = (unsigned short*)smem;                       // 64*256 bf16
  char*           region2 = smem + SMEM_ACTA;
  unsigned short* bufB    = (unsigned short*)region2;                    // 64*256 bf16
  float*          headbuf = (float*)region2;                             // 64*224 f32 (aliases bufB)
  unsigned short* wbuf    = (unsigned short*)(smem + SMEM_ACTA + SMEM_REG2); // staged weights

  const int tid  = threadIdx.x;
  const int lane = tid & 31;
  const int wave = tid >> 5;
  const int mt   = wave & 3;
  const int nh   = wave >> 2;
  const int base_pix = blockIdx.x * MROWS;

  // ---- kick off async staging of layer-0 weights; overlap with input build
  async_issue_weights(W0, wbuf, HID*KIN_PAD*2, tid);

  // ---- stage 1: build input rows [64][128] in bufB (feat | PE | frac | pad)
  {
    int r  = tid >> 2;
    int cb = (tid & 3) * 32;
    int pix = base_pix + r;
    int b   = pix >> 16;
    int rem = pix & 65535;
    int y = rem >> 8, x = rem & 255;
    float lx = (x + 0.5f)*0.25f - 0.5f;
    float ly = (y + 0.5f)*0.25f - 0.5f;
    int cx = iclamp((int)floorf(lx + 0.5f), 0, LRW-1);
    int cy = iclamp((int)floorf(ly + 0.5f), 0, LRH-1);
    const float* fv = feat + (((size_t)(b*LRH) + cy)*LRW + cx)*FEATC;
    float xn = (x + 0.5f)*(2.0f/HRW) - 1.0f;
    float yn = (y + 0.5f)*(2.0f/HRH) - 1.0f;
    for (int j = 0; j < 32; ++j) {
      int c = cb + j;
      float v = 0.f;
      if (c < 64) v = fv[c];
      else if (c < 104) {
        int q = c - 64;
        float fr = exp2f((float)(q >> 2)) * 3.14159265358979323846f;
        int comp = q & 3;
        float ang = (comp < 2 ? xn : yn) * fr;
        v = (comp & 1) ? __cosf(ang) : __sinf(ang);
      } else if (c == 104) v = lx - (float)cx;
      else if (c == 105) v = ly - (float)cy;
      bufB[r*HID + c] = f2bf(v);
    }
  }
  async_wait_all();
  __syncthreads();

  // ---- stage 2: 5 MLP layers + head GEMM; weights staged via async-to-LDS
  __builtin_prefetch(W1, 0, 3);
  gemm_layer<KIN_PAD>(bufB, wbuf, b0, bufA, lane, mt, nh);
  __syncthreads();
  async_issue_weights(W1, wbuf, HID*HID*2, tid); async_wait_all(); __syncthreads();
  __builtin_prefetch(W2, 0, 3);
  gemm_layer<HID>(bufA, wbuf, b1, bufB, lane, mt, nh);
  __syncthreads();
  async_issue_weights(W2, wbuf, HID*HID*2, tid); async_wait_all(); __syncthreads();
  __builtin_prefetch(W3, 0, 3);
  gemm_layer<HID>(bufB, wbuf, b2, bufA, lane, mt, nh);
  __syncthreads();
  async_issue_weights(W3, wbuf, HID*HID*2, tid); async_wait_all(); __syncthreads();
  __builtin_prefetch(W4, 0, 3);
  gemm_layer<HID>(bufA, wbuf, b3, bufB, lane, mt, nh);
  __syncthreads();
  async_issue_weights(W4, wbuf, HID*HID*2, tid); async_wait_all(); __syncthreads();
  __builtin_prefetch(WH, 0, 3);
  gemm_layer<HID>(bufB, wbuf, b4, bufA, lane, mt, nh);
  __syncthreads();
  async_issue_weights(WH, wbuf, NHEAD_PAD*HID*2, tid); async_wait_all(); __syncthreads();
  gemm_head(bufA, wbuf, bH, headbuf, lane, mt, nh);
  __syncthreads();

  // ---- stage 3: kernel-norm + gate softmax + 7x7x3 patch mixing
  if (tid < MROWS) {
    int r   = tid;
    int pix = base_pix + r;
    int b   = pix >> 16;
    int rem = pix & 65535;
    int y = rem >> 8, x = rem & 255;
    float lx = (x + 0.5f)*0.25f - 0.5f;
    float ly = (y + 0.5f)*0.25f - 0.5f;
    int cx = iclamp((int)floorf(lx + 0.5f), 0, LRW-1);
    int cy = iclamp((int)floorf(ly + 0.5f), 0, LRH-1);
    const float* hb = headbuf + r*NHEAD_PAD;

    float mu[4], sdiv[4];
#pragma unroll
    for (int h = 0; h < 4; ++h) {
      float s = 0.f, s2 = 0.f;
      for (int t = 0; t < 49; ++t) { float v = hb[h*49 + t]; s += v; s2 += v*v; }
      float m = s * (1.f/49.f);
      float var = fmaxf(s2 * (1.f/49.f) - m*m, 0.f);
      mu[h] = m;
      sdiv[h] = __builtin_amdgcn_rcpf(sqrtf(var) + 1e-6f);
    }
    float g[4], gm = -3.4e38f;
#pragma unroll
    for (int h = 0; h < 4; ++h) { g[h] = hb[196 + h]; gm = fmaxf(gm, g[h]); }
    float gs = 0.f;
#pragma unroll
    for (int h = 0; h < 4; ++h) { g[h] = __expf(g[h] - gm); gs += g[h]; }
    float inv_gs = __builtin_amdgcn_rcpf(gs);

    float m0 = 0.f, m1 = 0.f, m2 = 0.f;
    for (int t = 0; t < 49; ++t) {
      int oy = t / 7 - 3, ox = t % 7 - 3;
      int py = iclamp(cy + oy, 0, LRH-1);
      int px = iclamp(cx + ox, 0, LRW-1);
      float wsum = 0.f;
#pragma unroll
      for (int h = 0; h < 4; ++h)
        wsum += (g[h]*inv_gs) * ((hb[h*49 + t] - mu[h])*sdiv[h] + (1.f/49.f));
      const float* lp = lr + ((size_t)(b*3)*LRH + py)*LRW + px;
      m0 += wsum * lp[0];
      m1 += wsum * lp[LRH*LRW];
      m2 += wsum * lp[2*LRH*LRW];
    }
    size_t o = ((size_t)(b*3)*HRH + y)*HRW + x;
    res0[o]                     = m0;
    res0[o +   (size_t)HRH*HRW] = m1;
    res0[o + 2*(size_t)HRH*HRW] = m2;
  }
}

// ---------------------------------------------------------------------------
// Bicubic 4x upsample (Keys a=-0.5), NCHW
// ---------------------------------------------------------------------------
__device__ __forceinline__ float cubw(float t){
  const float a = -0.5f; t = fabsf(t);
  float w1 = ((a+2.f)*t - (a+3.f))*t*t + 1.f;        // |t|<=1
  float w2 = (((t-5.f)*t+8.f)*t-4.f)*a;              // 1<|t|<2
  float w  = (t <= 1.f) ? w1 : ((t < 2.f) ? w2 : 0.f);
  return w;
}
__global__ void k_bicubic(const float* __restrict__ lr, float* __restrict__ up)
{
  int idx = blockIdx.x*blockDim.x + threadIdx.x;
  if (idx >= NB*3*HRH*HRW) return;
  int x = idx & 255, y = (idx >> 8) & 255;
  int t = idx >> 16;
  int c = t % 3, b = t / 3;
  float sx = (x + 0.5f)*0.25f - 0.5f;
  float sy = (y + 0.5f)*0.25f - 0.5f;
  int ix = (int)floorf(sx), iy = (int)floorf(sy);
  float fx = sx - ix, fy = sy - iy;
  float wx[4], wy[4];
#pragma unroll
  for (int j = 0; j < 4; ++j) { wx[j] = cubw(fx - (float)(j-1)); wy[j] = cubw(fy - (float)(j-1)); }
  const float* src = lr + (size_t)(b*3 + c)*LRH*LRW;
  float s = 0.f;
#pragma unroll
  for (int jy = 0; jy < 4; ++jy) {
    int yy = iclamp(iy + jy - 1, 0, LRH-1);
#pragma unroll
    for (int jx = 0; jx < 4; ++jx) {
      int xx = iclamp(ix + jx - 1, 0, LRW-1);
      s += wy[jy]*wx[jx]*src[yy*LRW + xx];
    }
  }
  up[idx] = s;
}

// ---------------------------------------------------------------------------
// concat(lr_up, res0) -> [B][6][H][W]
// ---------------------------------------------------------------------------
__global__ void k_concat(const float* __restrict__ up, const float* __restrict__ r0,
                         float* __restrict__ out)
{
  int idx = blockIdx.x*blockDim.x + threadIdx.x;
  if (idx >= NB*6*HRH*HRW) return;
  int x = idx & 255, y = (idx >> 8) & 255;
  int t = idx >> 16;
  int ch = t % 6, b = t / 6;
  size_t src = ((size_t)(b*3 + (ch < 3 ? ch : ch-3))*HRH + y)*HRW + x;
  out[idx] = (ch < 3) ? up[src] : r0[src];
}

// ---------------------------------------------------------------------------
// Grouped 3x3 SAME conv (fp32), NCHW
// ---------------------------------------------------------------------------
__global__ void k_conv_ref(const float* __restrict__ in, const float* __restrict__ w,
                           const float* __restrict__ bias, float* __restrict__ out,
                           int cin_g, int cout, int groups, int do_gelu)
{
  int idx = blockIdx.x*blockDim.x + threadIdx.x;
  if (idx >= NB*cout*HRH*HRW) return;
  int x = idx & 255, y = (idx >> 8) & 255;
  int t = idx >> 16;
  int oc = t % cout, b = t / cout;
  int ocpg = cout / groups;
  int gidx = oc / ocpg;
  int cin_total = cin_g * groups;
  float s = bias[oc];
  for (int ky = 0; ky < 3; ++ky) {
    int iy = y + ky - 1;
    if (iy < 0 || iy >= HRH) continue;
    for (int kx = 0; kx < 3; ++kx) {
      int ix = x + kx - 1;
      if (ix < 0 || ix >= HRW) continue;
      const float* wp = w + (size_t)oc*cin_g*9 + ky*3 + kx;
      const float* ip = in + (((size_t)(b*cin_total + gidx*cin_g))*HRH + iy)*HRW + ix;
      for (int ic = 0; ic < cin_g; ++ic)
        s += ip[(size_t)ic*HRH*HRW] * wp[ic*9];
    }
  }
  out[idx] = do_gelu ? gelu_f(s) : s;
}

// ---------------------------------------------------------------------------
// Final: sr=clip(sr0+refine); out = clip(lr_up + (gray(sr)-gray(lr_up)))
// ---------------------------------------------------------------------------
__global__ void k_final(const float* __restrict__ up, const float* __restrict__ r0,
                        const float* __restrict__ rf, float* __restrict__ out)
{
  int idx = blockIdx.x*blockDim.x + threadIdx.x;
  if (idx >= NB*HRH*HRW) return;
  int x = idx & 255, y = (idx >> 8) & 255;
  int b = idx >> 16;
  float lu[3], sr[3];
#pragma unroll
  for (int c = 0; c < 3; ++c) {
    size_t o = ((size_t)(b*3 + c)*HRH + y)*HRW + x;
    lu[c] = up[o];
    sr[c] = fclampf(lu[c] + r0[o] + rf[o], 0.f, 1.f);
  }
  float gs = 0.2989f*sr[0] + 0.587f*sr[1] + 0.114f*sr[2];
  float gl = 0.2989f*lu[0] + 0.587f*lu[1] + 0.114f*lu[2];
  float d  = gs - gl;
#pragma unroll
  for (int c = 0; c < 3; ++c) {
    size_t o = ((size_t)(b*3 + c)*HRH + y)*HRW + x;
    out[o] = fclampf(lu[c] + d, 0.f, 1.f);
  }
}

// ---------------------------------------------------------------------------
// Launch
// ---------------------------------------------------------------------------
extern "C" void kernel_launch(void* const* d_in, const int* in_sizes, int n_in,
                              void* d_out, int out_size, void* d_ws, size_t ws_size,
                              hipStream_t stream)
{
  (void)in_sizes; (void)n_in; (void)out_size; (void)ws_size;
  const float* lr = (const float*)d_in[0];
  const float* enc_w[4] = {(const float*)d_in[1],(const float*)d_in[3],(const float*)d_in[5],(const float*)d_in[7]};
  const float* enc_b[4] = {(const float*)d_in[2],(const float*)d_in[4],(const float*)d_in[6],(const float*)d_in[8]};
  const float* mlp_w[5] = {(const float*)d_in[9],(const float*)d_in[11],(const float*)d_in[13],(const float*)d_in[15],(const float*)d_in[17]};
  const float* mlp_b[5] = {(const float*)d_in[10],(const float*)d_in[12],(const float*)d_in[14],(const float*)d_in[16],(const float*)d_in[18]};
  const float* hk_w = (const float*)d_in[19];
  const float* hk_b = (const float*)d_in[20];
  const float* hg_w = (const float*)d_in[21];
  const float* hg_b = (const float*)d_in[22];
  const float* ref_w[3] = {(const float*)d_in[23],(const float*)d_in[25],(const float*)d_in[27]};
  const float* ref_b[3] = {(const float*)d_in[24],(const float*)d_in[26],(const float*)d_in[28]};

  char* ws = (char*)d_ws;
  size_t off = 0;
  #define WS_ALLOC(type, nbytes) (type*)(ws + off); off += (((size_t)(nbytes)) + 255) & ~(size_t)255

  unsigned short* Wt0 = WS_ALLOC(unsigned short, HID*KIN_PAD*2);
  unsigned short* Wt1 = WS_ALLOC(unsigned short, HID*HID*2);
  unsigned short* Wt2 = WS_ALLOC(unsigned short, HID*HID*2);
  unsigned short* Wt3 = WS_ALLOC(unsigned short, HID*HID*2);
  unsigned short* Wt4 = WS_ALLOC(unsigned short, HID*HID*2);
  unsigned short* WtH = WS_ALLOC(unsigned short, NHEAD_PAD*HID*2);
  float* hbias = WS_ALLOC(float, NHEAD_PAD*4);
  float* featA = WS_ALLOC(float, (size_t)NB*LRH*LRW*FEATC*4);
  float* featB = WS_ALLOC(float, (size_t)NB*LRH*LRW*FEATC*4);
  float* lrup  = WS_ALLOC(float, (size_t)NB*3*HRH*HRW*4);
  float* res0  = WS_ALLOC(float, (size_t)NB*3*HRH*HRW*4);
  float* t0    = WS_ALLOC(float, (size_t)NB*6*HRH*HRW*4);
  float* t1    = WS_ALLOC(float, (size_t)NB*48*HRH*HRW*4);
  float* t2    = WS_ALLOC(float, (size_t)NB*48*HRH*HRW*4);
  float* rfout = WS_ALLOC(float, (size_t)NB*3*HRH*HRW*4);
  #undef WS_ALLOC

  // --- weight prep (bf16, transposed, padded) ---
  k_transpose_bf16<<<(HID*KIN_PAD+255)/256, 256, 0, stream>>>(mlp_w[0], Wt0, 106, HID, KIN_PAD);
  k_transpose_bf16<<<(HID*HID+255)/256, 256, 0, stream>>>(mlp_w[1], Wt1, HID, HID, HID);
  k_transpose_bf16<<<(HID*HID+255)/256, 256, 0, stream>>>(mlp_w[2], Wt2, HID, HID, HID);
  k_transpose_bf16<<<(HID*HID+255)/256, 256, 0, stream>>>(mlp_w[3], Wt3, HID, HID, HID);
  k_transpose_bf16<<<(HID*HID+255)/256, 256, 0, stream>>>(mlp_w[4], Wt4, HID, HID, HID);
  k_prep_head<<<(NHEAD_PAD*HID+255)/256, 256, 0, stream>>>(hk_w, hk_b, hg_w, hg_b, WtH, hbias);

  // --- encoder (4 conv layers) ---
  const int enc_n = NB*LRH*LRW*FEATC;
  k_conv_enc<<<(enc_n+255)/256, 256, 0, stream>>>(lr,    3,  1, enc_w[0], enc_b[0], featA);
  k_conv_enc<<<(enc_n+255)/256, 256, 0, stream>>>(featA, 64, 0, enc_w[1], enc_b[1], featB);
  k_conv_enc<<<(enc_n+255)/256, 256, 0, stream>>>(featB, 64, 0, enc_w[2], enc_b[2], featA);
  k_conv_enc<<<(enc_n+255)/256, 256, 0, stream>>>(featA, 64, 0, enc_w[3], enc_b[3], featB);

  // --- fused WMMA MLP + heads + patch mixing ---
  k_mlp_fused<<<(NB*HRH*HRW)/MROWS, 256, SMEM_BYTES, stream>>>(
      featB, lr,
      Wt0, mlp_b[0], Wt1, mlp_b[1], Wt2, mlp_b[2], Wt3, mlp_b[3], Wt4, mlp_b[4],
      WtH, hbias, res0);

  // --- bicubic upsample, concat, grouped refinement, final ---
  k_bicubic<<<(NB*3*HRH*HRW+255)/256, 256, 0, stream>>>(lr, lrup);
  k_concat<<<(NB*6*HRH*HRW+255)/256, 256, 0, stream>>>(lrup, res0, t0);
  k_conv_ref<<<(NB*48*HRH*HRW+255)/256, 256, 0, stream>>>(t0, ref_w[0], ref_b[0], t1,  2, 48, 3, 1);
  k_conv_ref<<<(NB*48*HRH*HRW+255)/256, 256, 0, stream>>>(t1, ref_w[1], ref_b[1], t2, 16, 48, 3, 1);
  k_conv_ref<<<(NB*3*HRH*HRW+255)/256, 256, 0, stream>>>(t2, ref_w[2], ref_b[2], rfout, 16, 3, 3, 0);
  k_final<<<(NB*HRH*HRW+255)/256, 256, 0, stream>>>(lrup, res0, rfout, (float*)d_out);
}
